// DLGN_VT_13030930776381
// MI455X (gfx1250) — compile-verified, matrix operand order checked by reading, not run on hardware
//
#include <hip/hip_runtime.h>
#include <hip/hip_bf16.h>

typedef __attribute__((ext_vector_type(16))) _Float16 v16h;
typedef __attribute__((ext_vector_type(8)))  _Float16 v8h;
typedef __attribute__((ext_vector_type(8)))  float    v8f;
typedef __attribute__((ext_vector_type(4)))  unsigned int u32x4;
typedef __attribute__((ext_vector_type(8)))  int          i32x8;
typedef __attribute__((ext_vector_type(4)))  int          i32x4;

#define BETA 30.0f

__device__ __forceinline__ float fast_sigmoid(float z) {
    return 1.0f / (1.0f + __expf(-z));
}

// ---------------------------------------------------------------------------
// TDM: 1D linear copy of `ndwords` dwords from global `gsrc` to LDS offset
// `lds_byte_off`. Descriptor per CDNA5 ISA §8.3-8.5:
//   group0: count=1 | lds_addr | global_addr | type=2
//   group1: data_size=4B, tensor_dim0=tile_dim0=ndwords, tensor_dim1=tile_dim1=1
// ---------------------------------------------------------------------------
__device__ __forceinline__ void tdm_load_1d_b4(const void* gsrc,
                                               unsigned lds_byte_off,
                                               unsigned ndwords) {
    unsigned long long ga = (unsigned long long)(uintptr_t)gsrc;
    u32x4 g0 = { 1u,                                        // count=1, user mode
                 lds_byte_off,                              // D#.lds_addr
                 (unsigned)(ga & 0xFFFFFFFFu),              // global_addr[31:0]
                 (unsigned)((ga >> 32) & 0x01FFFFFFu) | (2u << 30) }; // hi|type=2
    i32x8 g1;
    g1[0] = (int)(2u << 16);                                // data_size=4B
    g1[1] = (int)((ndwords & 0xFFFFu) << 16);               // tensor_dim0[15:0]
    g1[2] = (int)(((ndwords >> 16) & 0xFFFFu) | (1u << 16));// dim0[31:16]|dim1=1
    g1[3] = (int)((ndwords & 0xFFFFu) << 16);               // tile_dim0
    g1[4] = 1;                                              // tile_dim1=1
    g1[5] = (int)ndwords;                                   // dim0_stride[31:0]
    g1[6] = (int)((ndwords & 0xFFFFu) << 16);               // dim1_stride[15:0]
    g1[7] = (int)((ndwords >> 16) & 0xFFFFu);               // dim1_stride[47:16]
    i32x4 z4 = { 0, 0, 0, 0 };
#if defined(__clang_major__) && (__clang_major__ >= 23)
    i32x8 z8 = {};
    __builtin_amdgcn_tensor_load_to_lds(g0, g1, z4, z4, z8, 0);
#else
    __builtin_amdgcn_tensor_load_to_lds(g0, g1, z4, z4, 0);
#endif
}

// ---------------------------------------------------------------------------
// Prep kernel (one-off per launch): convert & swizzle everything to f16 in ws.
//  wsV [32768 halfs]: V in WMMA B-fragment order, idx=((nt*32+l)*16+e)
//  wsW [12288 halfs]: W fragments, idx=(((w*4+kb)*2+t)*32+l)*16+e
//  wsX [B*128 halfs]: x row-major f16
// ---------------------------------------------------------------------------
__global__ __launch_bounds__(256)
void dlgn_prep_kernel(const float* __restrict__ x,
                      const float* __restrict__ W1,
                      const float* __restrict__ W2,
                      const float* __restrict__ W3,
                      const float* __restrict__ V,
                      _Float16* __restrict__ wsV,
                      _Float16* __restrict__ wsW,
                      _Float16* __restrict__ wsX,
                      int Btotal) {
    int t = blockIdx.x * 256 + threadIdx.x;
    if (t < 32768) {                       // ---- V swizzle ----
        int e = t & 15, l = (t >> 4) & 31, nt = t >> 9;
        int bh = l >> 4, n = nt * 16 + (l & 15);
        int i = n >> 5, j = n & 31;
        int k = (e < 8) ? (8 * bh + e) : (16 + 8 * bh + (e - 8));
        wsV[t] = (_Float16)V[(i * 32 + j) * 32 + k];
    } else if (t < 32768 + 12288) {        // ---- W swizzle ----
        int q = t - 32768;
        int e = q & 15, l = (q >> 4) & 31;
        int tt = (q >> 9) & 1, kb = (q >> 10) & 3, w = q >> 12;
        int bh = l >> 4, col = tt * 16 + (l & 15);
        int k = 32 * kb + ((e < 8) ? (8 * bh + e) : (16 + 8 * bh + (e - 8)));
        const float* Wp = (w == 0) ? W1 : ((w == 1) ? W2 : W3);
        wsW[q] = (_Float16)Wp[col * 128 + k];
    } else {                               // ---- x convert (float4/thread) ----
        int q = t - 45056;
        if (q * 4 < Btotal * 128) {
            float4 v = ((const float4*)x)[q];
            wsX[q * 4 + 0] = (_Float16)v.x;
            wsX[q * 4 + 1] = (_Float16)v.y;
            wsX[q * 4 + 2] = (_Float16)v.z;
            wsX[q * 4 + 3] = (_Float16)v.w;
        }
    }
}

// ---------------------------------------------------------------------------
// Main fused kernel: grid=B/64 blocks, 128 threads (4 waves), 16 rows/wave.
// LDS: [0,65536) swizzled V (filled by TDM), [65536,...) per-wave G3 staging.
// ---------------------------------------------------------------------------
__global__ __launch_bounds__(128)
void dlgn_fused_kernel(const _Float16* __restrict__ wsV,
                       const _Float16* __restrict__ wsW,
                       const _Float16* __restrict__ wsX,
                       float* __restrict__ out) {
    extern __shared__ char smem_raw[];
    _Float16* smB   = (_Float16*)smem_raw;           // 64 KB swizzled V
    float*    stage = (float*)(smem_raw + 65536);    // 4 waves * 16*33 floats

    const int tid  = threadIdx.x;
    const int lane = tid & 31;
    const int wave = __builtin_amdgcn_readfirstlane(tid >> 5);
    const int nl   = lane & 15;
    const int h    = lane >> 4;

    // ---- Stage 0: async DMA of this wave's 16 KB quarter of V into LDS ----
    tdm_load_1d_b4((const char*)wsV + wave * 16384, (unsigned)(wave * 16384),
                   4096u);

    // ---- Stage 1: gate GEMMs  G_w = sigmoid(30 * x @ W_w^T) ----
    const int rowBase = blockIdx.x * 64 + wave * 16;
    const _Float16* xh = wsX + (size_t)(rowBase + nl) * 128;
    __builtin_prefetch(xh, 0, 3);   // global_prefetch_b8

    v8f zero8 = {};
    v8f accg[3][2];
#pragma unroll
    for (int w = 0; w < 3; ++w)
#pragma unroll
        for (int t = 0; t < 2; ++t) accg[w][t] = zero8;

#pragma unroll
    for (int kb = 0; kb < 4; ++kb) {
        // A fragment: row = nl, K-chunks 8h..8h+7 and 16+8h..16+8h+7.
        v8h lo = *(const v8h*)(xh + 32 * kb + 8 * h);
        v8h hi = *(const v8h*)(xh + 32 * kb + 16 + 8 * h);
        v16h a;
#pragma unroll
        for (int e = 0; e < 8; ++e) { a[e] = lo[e]; a[8 + e] = hi[e]; }
#pragma unroll
        for (int w = 0; w < 3; ++w) {
#pragma unroll
            for (int t = 0; t < 2; ++t) {
                const v16h b = *(const v16h*)(wsW +
                    (size_t)((((w * 4 + kb) * 2 + t) * 32 + lane) * 16));
                accg[w][t] = __builtin_amdgcn_wmma_f32_16x16x32_f16(
                    false, a, false, b, (short)0, accg[w][t], false, false);
            }
        }
    }

    // Sigmoid; G1/G2 stay in registers, G3 staged to LDS for A-relayout.
    float* st = stage + wave * (16 * 33);
    v8f g1f[2], g2f[2];
#pragma unroll
    for (int t = 0; t < 2; ++t) {
#pragma unroll
        for (int r = 0; r < 8; ++r) {
            g1f[t][r] = fast_sigmoid(BETA * accg[0][t][r]);
            g2f[t][r] = fast_sigmoid(BETA * accg[1][t][r]);
            float g3  = fast_sigmoid(BETA * accg[2][t][r]);
            st[(r + 8 * h) * 33 + t * 16 + nl] = g3;
        }
    }

    __builtin_amdgcn_s_wait_tensorcnt(0);   // our TDM quarter landed
    __syncthreads();                        // all quarters + all staging

    // Rebuild G3 as WMMA A fragment (row = nl).
    v16h a3;
    {
        const float* srow = st + nl * 33;
#pragma unroll
        for (int e = 0; e < 8; ++e) {
            a3[e]     = (_Float16)srow[8 * h + e];
            a3[8 + e] = (_Float16)srow[16 + 8 * h + e];
        }
    }

    // ---- Stage 2: contraction T = G3 @ Vr fused with g1*g2 weighting ----
    // Tile nt: i = nt>>1 (const per tile), j = 16*(nt&1) + nl
    //   -> g2 multiplier is our own G2 fragment (same lane/VGPR)
    //   -> g1 multiplier is a 16-lane broadcast from lane (i&15)|16h
    v8f acc = zero8;
    for (int nt = 0; nt < 64; ++nt) {
        const v16h bf = *(const v16h*)(smB + nt * 512 + lane * 16);
        v8f c0 = {};
        v8f tC = __builtin_amdgcn_wmma_f32_16x16x32_f16(
            false, a3, false, bf, (short)0, c0, false, false);

        const v8f g2sel = (nt & 1) ? g2f[1] : g2f[0];
        const int i = nt >> 1;
        const v8f g1sel = (i >> 4) ? g1f[1] : g1f[0];
        const int srcLane = (i & 15) | (lane & 16);
#pragma unroll
        for (int r = 0; r < 8; ++r) {
            float g1v = __shfl(g1sel[r], srcLane, 32);
            acc[r] += tC[r] * g2sel[r] * g1v;
        }
    }

    // ---- Stage 3: per-row reduction over 16 lanes, store ----
#pragma unroll
    for (int r = 0; r < 8; ++r) {
        float v = acc[r];
        v += __shfl_xor(v, 1, 32);
        v += __shfl_xor(v, 2, 32);
        v += __shfl_xor(v, 4, 32);
        v += __shfl_xor(v, 8, 32);
        acc[r] = v;
    }
    if (nl == 0) {
#pragma unroll
        for (int r = 0; r < 8; ++r)
            out[rowBase + 8 * h + r] = acc[r];
    }
}

extern "C" void kernel_launch(void* const* d_in, const int* in_sizes, int n_in,
                              void* d_out, int out_size, void* d_ws, size_t ws_size,
                              hipStream_t stream) {
    const float* x  = (const float*)d_in[0];
    const float* W1 = (const float*)d_in[1];
    const float* W2 = (const float*)d_in[2];
    const float* W3 = (const float*)d_in[3];
    const float* V  = (const float*)d_in[4];
    float* out = (float*)d_out;

    const int B = in_sizes[0] / 128;   // 4096

    _Float16* wsV = (_Float16*)d_ws;           // 32768 halfs = 64 KB
    _Float16* wsW = wsV + 32768;               // 12288 halfs = 24 KB
    _Float16* wsX = wsW + 12288;               // B*128 halfs = 1 MB

    int prepThreads = 45056 + B * 32;          // V + W + x(float4)
    dim3 pgrid((unsigned)((prepThreads + 255) / 256)), pblock(256);
    hipLaunchKernelGGL(dlgn_prep_kernel, pgrid, pblock, 0, stream,
                       x, W1, W2, W3, V, wsV, wsW, wsX, B);

    dim3 grid((unsigned)(B / 64)), block(128);
    size_t shmem = 65536 + 4 * 16 * 33 * sizeof(float);   // 73,984 B
    hipLaunchKernelGGL(dlgn_fused_kernel, grid, block, shmem, stream,
                       wsV, wsW, wsX, out);
}
